// BoundaryPredictor2_88648124991229
// MI455X (gfx1250) — matrix operand order repaired and checked by reading, not compiled, not run on hardware
//
#include <hip/hip_runtime.h>
#include <hip/hip_bf16.h>
#include <math.h>

// ---------------------------------------------------------------------------
// BoundaryPredictor2 for MI455X (gfx1250, wave32, WMMA + TDM)
// ---------------------------------------------------------------------------

typedef __attribute__((ext_vector_type(16))) __bf16         v16bf;
typedef __attribute__((ext_vector_type(8)))  float          v8f;
typedef __attribute__((ext_vector_type(4)))  float          f32x4;
typedef __attribute__((ext_vector_type(4)))  unsigned int   u32x4;
typedef __attribute__((ext_vector_type(8)))  int            i32x8;
typedef __attribute__((ext_vector_type(4)))  int            i32x4;
typedef __attribute__((ext_vector_type(4)))  unsigned short u16x4;

static constexpr int kB = 8;
static constexpr int kL = 4096;
static constexpr int kD = 512;
static constexpr float kPrior = 0.2f;

__device__ __forceinline__ unsigned short f2bf(float f) {
    unsigned u = __float_as_uint(f);
    u += 0x7FFFu + ((u >> 16) & 1u);   // round-to-nearest-even
    return (unsigned short)(u >> 16);
}

#define LDS_AS __attribute__((address_space(3)))
__device__ __forceinline__ unsigned ldsByteOffset(const void* p) {
    // raw LDS byte address for TDM descriptors (AS(3) ptrtoint)
    return (unsigned)(unsigned long long)(LDS_AS const void*)p;
}

#if __has_builtin(__builtin_amdgcn_tensor_load_to_lds)
#define HAVE_TDM 1
// TDM: async-load a 16-row x 512-col bf16 tile (row stride 512 elems) to LDS.
// D# per CDNA5 ISA (2D tensor, groups 2/3 zero).
// clang-23 lane: 6-arg builtin (g0, g1, g2, g3, g_extra, cpol).
__device__ __forceinline__ void tdm_load_tile16(const unsigned short* gtile,
                                                unsigned ldsOff) {
    unsigned long long ga = (unsigned long long)gtile;
    u32x4 g0;
    g0.x = 1u;                                   // count=1, user descriptor
    g0.y = ldsOff;                               // lds_addr (bytes)
    g0.z = (unsigned)(ga & 0xFFFFFFFFu);         // global_addr[31:0]
    g0.w = (unsigned)((ga >> 32) & 0x01FFFFFFu)  // global_addr[56:32]
         | (2u << 30);                           // type = 2 ("image")
    i32x8 g1;
    g1[0] = (1 << 16);                 // data_size=1 (2 bytes); mask/flags = 0
    g1[1] = (int)((unsigned)kD << 16); // tensor_dim0[15:0]  in [31:16]
    g1[2] = (int)((unsigned)kD << 16); // dim0[31:16]=0 | tensor_dim1[15:0]
    g1[3] = (int)((unsigned)kD << 16); // dim1[31:16]=0 | tile_dim0=512
    g1[4] = 16;                        // tile_dim1=16, tile_dim2=0
    g1[5] = kD;                        // tensor_dim0_stride = 512 elems
    g1[6] = 0;                         // stride hi | dim1_stride (unused 2D)
    g1[7] = 0;
    i32x4 z4 = {0, 0, 0, 0};
    i32x8 z8 = {0, 0, 0, 0, 0, 0, 0, 0};
    __builtin_amdgcn_tensor_load_to_lds(g0, g1, z4, z4, z8, 0);
}
#else
#define HAVE_TDM 0
#endif

// --- 1) convert q_w / k_w to bf16 ------------------------------------------
__global__ void k_convert_weights(const float* __restrict__ qw,
                                  const float* __restrict__ kw,
                                  unsigned short* __restrict__ qwb,
                                  unsigned short* __restrict__ kwb) {
    int idx = blockIdx.x * blockDim.x + threadIdx.x;
    const int n = kD * kD;
    if (idx < n)            qwb[idx]     = f2bf(qw[idx]);
    else if (idx < 2 * n)   kwb[idx - n] = f2bf(kw[idx - n]);
}

// --- 2) L2-normalize each token row, emit bf16 -----------------------------
__global__ void k_normalize(const float* __restrict__ hidden,
                            unsigned short* __restrict__ nbf) {
    const int row  = blockIdx.x;                 // 0 .. B*L-1
    const int t    = threadIdx.x;                // 128 threads, 4 floats each
    const int lane = t & 31, warp = t >> 5;
    const f32x4* h4 = (const f32x4*)(hidden + (size_t)row * kD);
    f32x4 v = h4[t];
    float ss = v.x * v.x + v.y * v.y + v.z * v.z + v.w * v.w;
    #pragma unroll
    for (int off = 16; off > 0; off >>= 1) ss += __shfl_xor(ss, off);
    __shared__ float red[4];
    if (lane == 0) red[warp] = ss;
    __syncthreads();
    float tot   = red[0] + red[1] + red[2] + red[3];
    float scale = 1.0f / fmaxf(sqrtf(tot), 1e-12f);
    u16x4 o;
    o.x = f2bf(v.x * scale); o.y = f2bf(v.y * scale);
    o.z = f2bf(v.z * scale); o.w = f2bf(v.w * scale);
    ((u16x4*)(nbf + (size_t)row * kD))[t] = o;
}

// --- 3) fused bf16 WMMA GEMM: Q = Nbf @ qw^T, K = Nbf @ kw^T ---------------
// Block: 256 threads = 8 waves; block covers 128 rows x 16 cols for BOTH
// projections (A loaded once, two WMMAs per K-step). Weight tiles (2 x 16 KB)
// staged in LDS by the Tensor Data Mover (wave 0 issues, TENSORcnt-tracked).
__launch_bounds__(256)
__global__ void k_gemm_qk(const unsigned short* __restrict__ nbf,
                          const unsigned short* __restrict__ qwb,
                          const unsigned short* __restrict__ kwb,
                          float* __restrict__ Q,
                          float* __restrict__ K) {
    __shared__ alignas(32) unsigned short qTile[16][kD];   // 16 KB
    __shared__ alignas(32) unsigned short kTile[16][kD];   // 16 KB

    const int t  = threadIdx.x;
    const int n0 = blockIdx.y * 16;

#if HAVE_TDM
    if (t < 32) {                       // wave 0 drives the TDM (EXEC ignored)
        tdm_load_tile16(qwb + (size_t)n0 * kD, ldsByteOffset(&qTile[0][0]));
        tdm_load_tile16(kwb + (size_t)n0 * kD, ldsByteOffset(&kTile[0][0]));
        __builtin_amdgcn_s_wait_tensorcnt(0);
    }
#else
    // cooperative fallback: 2048 x 16-byte chunks across 256 threads
    {
        const u32x4* qs = (const u32x4*)qwb;
        const u32x4* ks = (const u32x4*)kwb;
        u32x4* qd = (u32x4*)&qTile[0][0];
        u32x4* kd = (u32x4*)&kTile[0][0];
        #pragma unroll
        for (int i = 0; i < 4; ++i) {
            int idx = t + i * 256;                 // 0..1023
            int r = idx >> 6, c = idx & 63;
            qd[idx] = qs[(size_t)(n0 + r) * 64 + c];
            kd[idx] = ks[(size_t)(n0 + r) * 64 + c];
        }
    }
#endif
    __syncthreads();

    const int wave  = t >> 5;
    const int lane  = t & 31;
    const int laneM = lane & 15;        // A row index / B,C column index
    const int lh    = lane >> 4;        // half-wave select

    const int m0 = blockIdx.x * 128 + wave * 16;
    const u32x4* arow = (const u32x4*)(nbf + (size_t)(m0 + laneM) * kD);

    v8f accQ = {};
    v8f accK = {};
    #pragma unroll
    for (int k0 = 0; k0 < kD; k0 += 32) {
        if (k0 + 64 < kD) __builtin_prefetch(arow + (k0 >> 3) + 8, 0, 1);
        // A (16x32 bf16): lane holds K = k0+16*lh+{0..15}
        union { u32x4 q[2]; v16bf v; } ac;
        ac.q[0] = arow[(k0 >> 3) + 2 * lh];
        ac.q[1] = arow[(k0 >> 3) + 2 * lh + 1];
        // B (32x16 bf16) = W^T : lane holds W[n0+laneM][k0+16*lh .. +16]
        const v16bf bq = *(const v16bf*)&qTile[laneM][k0 + 16 * lh];
        const v16bf bk = *(const v16bf*)&kTile[laneM][k0 + 16 * lh];
        accQ = __builtin_amdgcn_wmma_f32_16x16x32_bf16(
                   false, ac.v, false, bq, (short)0, accQ, false, false);
        accK = __builtin_amdgcn_wmma_f32_16x16x32_bf16(
                   false, ac.v, false, bk, (short)0, accK, false, false);
    }

    // C/D layout: VGPR r -> M = r + 8*lh, N = laneM
    #pragma unroll
    for (int r = 0; r < 8; ++r) {
        size_t o = (size_t)(m0 + r + 8 * lh) * kD + n0 + laneM;
        Q[o] = accQ[r];
        K[o] = accK[r];
    }
}

// --- 4) boundary prob -> RelaxedBernoulli sample -> hard threshold ---------
// One wave per (b,l).  hard[b,0] = 1 (padded prob 1 -> logits=+inf).
__global__ void k_boundary(const float* __restrict__ Q,
                           const float* __restrict__ K,
                           const float* __restrict__ noise,
                           float* __restrict__ hard) {
    const int wave = threadIdx.x >> 5, lane = threadIdx.x & 31;
    const int p = blockIdx.x * 8 + wave;            // flat (b,l)
    const int l = p & (kL - 1);
    if (l == 0) { if (lane == 0) hard[p] = 1.0f; return; }
    const f32x4* q4 = (const f32x4*)(Q + (size_t)(p - 1) * kD);  // row (b,l-1)
    const f32x4* k4 = (const f32x4*)(K + (size_t)p * kD);        // row (b,l)
    float s = 0.f;
    #pragma unroll
    for (int i = 0; i < 4; ++i) {
        f32x4 a = q4[lane + 32 * i], b = k4[lane + 32 * i];
        s += a.x * b.x + a.y * b.y + a.z * b.z + a.w * b.w;
    }
    #pragma unroll
    for (int off = 16; off > 0; off >>= 1) s += __shfl_xor(s, off);
    if (lane == 0) {
        float prob   = fminf(fmaxf((1.0f - s) * 0.5f, 0.0f), 1.0f);
        float logits = logf(prob) - log1pf(-prob);  // +/-inf at ends, never NaN
        float nz     = noise[p];
        float lno    = logf(nz) - log1pf(-nz);
        float x      = logits + lno;                // TEMP = 1
        float soft   = 1.0f / (1.0f + expf(-x));
        hard[p] = (soft > 0.5f) ? 1.0f : 0.0f;
    }
}

// --- 5) per-batch scan: segment starts + boundary count --------------------
// seg = clip(cumsum(hard)-1, 0, L-1) is monotone: segments are contiguous.
__global__ void k_scan(const float* __restrict__ hard,
                       int* __restrict__ segStart,
                       float* __restrict__ ksum) {
    const int b = blockIdx.x, t = threadIdx.x;    // 128 threads, 32 elems each
    const int CH = kL / 128;
    __shared__ float wsum[128];
    __shared__ int   segRow[kL];                  // 16 KB

    for (int i = 0; i < CH; ++i) segStart[b * kL + t * CH + i] = -1;

    float local = 0.f;
    for (int i = 0; i < CH; ++i) local += hard[b * kL + t * CH + i];
    wsum[t] = local;
    __syncthreads();
    if (t == 0) {
        float run = 0.f;
        for (int i = 0; i < 128; ++i) { float x = wsum[i]; wsum[i] = run; run += x; }
        ksum[b] = run;                            // total #boundaries
    }
    __syncthreads();
    float run = wsum[t];
    for (int i = 0; i < CH; ++i) {
        int l = t * CH + i;
        run += hard[b * kL + l];
        int s = (int)(run + 0.5f) - 1;
        s = max(0, min(s, kL - 1));
        segRow[l] = s;
    }
    __syncthreads();
    for (int i = 0; i < CH; ++i) {
        int l = t * CH + i;
        int s = segRow[l];
        if (l == 0 || segRow[l - 1] != s) segStart[b * kL + s] = l;
    }
}

// --- 6) segment mean-pool into output (L, B, D) ----------------------------
__global__ void k_pool(const float* __restrict__ hidden,
                       const int* __restrict__ segStart,
                       float* __restrict__ out) {
    const int s = blockIdx.x, b = blockIdx.y, t = threadIdx.x;  // 128 thr x 4
    float* dst = out + ((size_t)s * kB + b) * kD;
    int st = segStart[b * kL + s];
    if (st < 0) { f32x4 z = {0.f, 0.f, 0.f, 0.f}; ((f32x4*)dst)[t] = z; return; }
    int en;
    if (s + 1 < kL) { int n1 = segStart[b * kL + s + 1]; en = (n1 < 0) ? kL : n1; }
    else            en = kL;
    f32x4 acc = {0.f, 0.f, 0.f, 0.f};
    for (int r = st; r < en; ++r)
        acc += ((const f32x4*)(hidden + ((size_t)b * kL + r) * kD))[t];
    float inv = 1.0f / (float)(en - st);
    ((f32x4*)dst)[t] = acc * inv;
}

// --- 7) Binomial(n=L, p=prior) log-prob loss -------------------------------
__global__ void k_loss(const float* __restrict__ ksum, float* __restrict__ loss) {
    const int t = threadIdx.x;                    // 32 threads
    const float n = (float)kL;
    float term = 0.f;
    if (t < kB) {
        float k = ksum[t];
        term = lgammaf(n + 1.0f) - lgammaf(k + 1.0f) - lgammaf(n - k + 1.0f)
             + k * logf(kPrior) + (n - k) * log1pf(-kPrior);
    }
    #pragma unroll
    for (int off = 16; off > 0; off >>= 1) term += __shfl_xor(term, off);
    if (t == 0) *loss = -(term / (float)kB) / n;
}

// ---------------------------------------------------------------------------
extern "C" void kernel_launch(void* const* d_in, const int* in_sizes, int n_in,
                              void* d_out, int out_size, void* d_ws, size_t ws_size,
                              hipStream_t stream) {
    (void)in_sizes; (void)n_in; (void)out_size; (void)ws_size;
    const float* hidden = (const float*)d_in[0];   // (8,4096,512) f32
    const float* q_w    = (const float*)d_in[1];   // (512,512)    f32
    const float* k_w    = (const float*)d_in[2];   // (512,512)    f32
    const float* noise  = (const float*)d_in[3];   // (8,4096)     f32
    float*       out    = (float*)d_out;           // pooled (L,B,D) ++ loss

    // workspace carve-up (~162 MiB)
    char* w = (char*)d_ws;
    unsigned short* Nbf = (unsigned short*)w; w += (size_t)kB * kL * kD * 2;
    unsigned short* qwb = (unsigned short*)w; w += (size_t)kD * kD * 2;
    unsigned short* kwb = (unsigned short*)w; w += (size_t)kD * kD * 2;
    float* Q    = (float*)w; w += (size_t)kB * kL * kD * 4;
    float* Kk   = (float*)w; w += (size_t)kB * kL * kD * 4;
    float* hard = (float*)w; w += (size_t)kB * kL * 4;
    int*   segS = (int*)w;   w += (size_t)kB * kL * 4;
    float* ksum = (float*)w;

    k_convert_weights<<<(2 * kD * kD) / 256, 256, 0, stream>>>(q_w, k_w, qwb, kwb);
    k_normalize<<<kB * kL, 128, 0, stream>>>(hidden, Nbf);

    dim3 gGemm((kB * kL) / 128, kD / 16);
    k_gemm_qk<<<gGemm, 256, 0, stream>>>(Nbf, qwb, kwb, Q, Kk);

    k_boundary<<<(kB * kL) / 8, 256, 0, stream>>>(Q, Kk, noise, hard);
    k_scan<<<kB, 128, 0, stream>>>(hard, segS, ksum);

    dim3 gPool(kL, kB);
    k_pool<<<gPool, 128, 0, stream>>>(hidden, segS, out);
    k_loss<<<1, 32, 0, stream>>>(ksum, out + (size_t)kL * kB * kD);
}
